// STConvBlock_53558242181187
// MI455X (gfx1250) — matrix operand order repaired and checked by reading, not compiled, not run on hardware
//
#include <hip/hip_runtime.h>
#include <hip/hip_fp16.h>
#include <math.h>

typedef __attribute__((ext_vector_type(16))) _Float16 v16h;
typedef __attribute__((ext_vector_type(8)))  _Float16 v8h;
typedef __attribute__((ext_vector_type(8)))  float    v8f;

#define BB 32
#define NN 1024
#define CT 64
#define CS 16
#define T1 22   // after TCN1
#define T2 20   // after TCN2

// ---------------------------------------------------------------------------
// Stage 0: cheb [3][m=1024][n=1024] f32  ->  chebT [3][n][m] f16 (A operand)
// ---------------------------------------------------------------------------
__global__ __launch_bounds__(256) void k_chebT(const float* __restrict__ cheb,
                                               _Float16* __restrict__ chebT) {
  int idx = blockIdx.x * 256 + threadIdx.x;      // 3*1024*1024 exact
  int m = idx & (NN - 1);
  int n = (idx >> 10) & (NN - 1);
  int k = idx >> 20;
  chebT[((size_t)k << 20) | ((size_t)n << 10) | (size_t)m] =
      (_Float16)cheb[((size_t)k << 20) | ((size_t)m << 10) | (size_t)n];
}

// ---------------------------------------------------------------------------
// Stage 1: gated TCN1 (1 -> 64ch) fused with  @ w_cheb[k] (64 -> 16ch, x3 k)
// writes hWt[k][b][t][o=16][m=1024] as f16 (WMMA B operand, contiguous in m)
// ---------------------------------------------------------------------------
__global__ __launch_bounds__(256) void k_tcn1(
    const float* __restrict__ x,                 // [32][24][1024]
    const float* __restrict__ w1, const float* __restrict__ b1,
    const float* __restrict__ w2, const float* __restrict__ b2,
    const float* __restrict__ w3, const float* __restrict__ b3,
    const float* __restrict__ wch,               // [3][64][16]
    _Float16* __restrict__ hWt)                  // [3][32][22][16][1024]
{
  __shared__ float sw1[CT*3], sw2[CT*3], sw3[CT*3];
  __shared__ float sb1[CT], sb2[CT], sb3[CT];
  __shared__ float swc[3*CT*CS];
  int tid = threadIdx.x;
  for (int i = tid; i < CT*3; i += 256) { sw1[i]=w1[i]; sw2[i]=w2[i]; sw3[i]=w3[i]; }
  if (tid < CT) { sb1[tid]=b1[tid]; sb2[tid]=b2[tid]; sb3[tid]=b3[tid]; }
  for (int i = tid; i < 3*CT*CS; i += 256) swc[i] = wch[i];
  __syncthreads();

  int m = blockIdx.x * 256 + tid;                // node
  int t = blockIdx.y;                            // 0..21
  int b = blockIdx.z;                            // 0..31
  const float* xp = x + ((size_t)b*24 + t)*NN + m;
  float x0 = xp[0], x1 = xp[NN], x2 = xp[2*NN];

  float acc[3*CS];
  #pragma unroll
  for (int i = 0; i < 3*CS; ++i) acc[i] = 0.f;

  for (int c = 0; c < CT; ++c) {
    float p = sb1[c] + sw1[c*3]*x0 + sw1[c*3+1]*x1 + sw1[c*3+2]*x2;
    float q = sb2[c] + sw2[c*3]*x0 + sw2[c*3+1]*x1 + sw2[c*3+2]*x2;
    float r = sb3[c] + sw3[c*3]*x0 + sw3[c*3+1]*x1 + sw3[c*3+2]*x2;
    float h = p * (1.f / (1.f + __expf(-q))) + tanhf(r);
    #pragma unroll
    for (int k = 0; k < 3; ++k) {
      #pragma unroll
      for (int o = 0; o < CS; ++o)
        acc[k*CS + o] += h * swc[(k*CT + c)*CS + o];
    }
  }
  #pragma unroll
  for (int k = 0; k < 3; ++k) {
    #pragma unroll
    for (int o = 0; o < CS; ++o)
      hWt[((((size_t)k*BB + b)*T1 + t)*CS + o)*NN + m] = (_Float16)acc[k*CS + o];
  }
}

// ---------------------------------------------------------------------------
// Stage 2: graph conv GEMM via WMMA f16.
// s[b,t][n,o] = sum_k chebT[k][n,:] . hWt[k][b,t][o,:]   (K = 1024 per k)
// One wave -> 4 n-tiles x 2 (b,t) slices = 8 accumulator tiles.
//   A fragments (chebT) reused across the 2 bt slices,
//   B fragments (hW) reused across the 4 n-tiles:
//   per K-step: 8 A-loads + 4 B-loads -> 8 WMMAs (1.5 b128 loads / wmma).
// All loads of a K-step are issued before the 8 WMMAs to keep the XDL pipe
// fed and let the scheduler hoist next-step loads (loadcnt overlap).
// ---------------------------------------------------------------------------
__global__ __launch_bounds__(256) void k_cheb_wmma(
    const _Float16* __restrict__ chebT,          // [3][1024][1024]
    const _Float16* __restrict__ hWt,            // [3][32][22][16][1024]
    const float*    __restrict__ bch,            // [16]
    float*          __restrict__ s)              // [32][22][1024][16]
{
  int wave = threadIdx.x >> 5;
  int lane = threadIdx.x & 31;
  int tg  = blockIdx.x * 8 + wave;               // tile-group 0..15 (4 tiles)
  int btp = blockIdx.y;                          // 0..351 (pair of bt slices)

  int row = lane & 15;                           // A row in tile / B column (o)
  int blk = lane >> 4;                           // lane-block 0/1

  const _Float16* Abase = chebT + (size_t)(tg*64 + row) * NN; // tile i at +i*16*NN
  const size_t Astride_k = (size_t)NN * NN;                   // next cheb tap
  const size_t Bstride_k = (size_t)BB * T1 * CS * NN;         // next k in hWt
  const size_t Atile     = (size_t)16 * NN;                   // next n-tile

  int btidx[2]; btidx[0] = btp*2; btidx[1] = btp*2 + 1;
  const _Float16* Bbase[2];
  #pragma unroll
  for (int j = 0; j < 2; ++j) {
    int b = btidx[j] / T1, t = btidx[j] % T1;
    Bbase[j] = hWt + ((((size_t)b)*T1 + t)*CS + row) * NN;
  }

  v8f acc[8];                                    // [bt j][tile]
  #pragma unroll
  for (int i = 0; i < 8; ++i) acc[i] = (v8f){};

  for (int k = 0; k < 3; ++k) {
    const _Float16* A = Abase + (size_t)k * Astride_k;
    const _Float16* B0 = Bbase[0] + (size_t)k * Bstride_k;
    const _Float16* B1 = Bbase[1] + (size_t)k * Bstride_k;
    for (int kt = 0; kt < 32; ++kt) {
      int kk = kt * 32;
      // ---- issue ALL loads for this K-step first ----
      // B fragments (32x16 f16): lane column o=row, K in [blk*8, blk*8+7] and [16+blk*8, 16+blk*8+7]
      v8h b00 = *reinterpret_cast<const v8h*>(B0 + kk + blk*8);
      v8h b01 = *reinterpret_cast<const v8h*>(B0 + kk + 16 + blk*8);
      v8h b10 = *reinterpret_cast<const v8h*>(B1 + kk + blk*8);
      v8h b11 = *reinterpret_cast<const v8h*>(B1 + kk + 16 + blk*8);
      // A fragments (16x32 f16): lane-block interleave -> two 16B loads
      v8h a0[4], a1[4];
      #pragma unroll
      for (int tile = 0; tile < 4; ++tile) {
        const _Float16* At = A + (size_t)tile * Atile + kk;
        a0[tile] = *reinterpret_cast<const v8h*>(At + blk*8);
        a1[tile] = *reinterpret_cast<const v8h*>(At + 16 + blk*8);
      }
      // ---- pack fragments ----
      v16h bv[2], av[4];
      #pragma unroll
      for (int i = 0; i < 8; ++i) { bv[0][i] = b00[i]; bv[0][i+8] = b01[i]; }
      #pragma unroll
      for (int i = 0; i < 8; ++i) { bv[1][i] = b10[i]; bv[1][i+8] = b11[i]; }
      #pragma unroll
      for (int tile = 0; tile < 4; ++tile) {
        #pragma unroll
        for (int i = 0; i < 8; ++i) { av[tile][i] = a0[tile][i]; av[tile][i+8] = a1[tile][i]; }
      }
      // ---- 8 back-to-back WMMAs ----
      #pragma unroll
      for (int j = 0; j < 2; ++j) {
        #pragma unroll
        for (int tile = 0; tile < 4; ++tile) {
          acc[j*4 + tile] = __builtin_amdgcn_wmma_f32_16x16x32_f16(
              /*neg_a=*/false, av[tile], /*neg_b=*/false, bv[j],
              /*c_mod=*/(short)0, acc[j*4 + tile],
              /*reuse_a=*/false, /*reuse_b=*/false);
        }
      }
    }
  }

  // D layout: lanes 0-15 -> M=r, lanes 16-31 -> M=8+r; column = lane&15
  int col = row;
  float bias = bch[col];
  int roff = blk * 8;
  #pragma unroll
  for (int j = 0; j < 2; ++j) {
    int b = btidx[j] / T1, t = btidx[j] % T1;
    size_t out0 = (((size_t)b*T1 + t)*NN + (size_t)tg*64) * CS;
    #pragma unroll
    for (int tile = 0; tile < 4; ++tile) {
      size_t ob = out0 + (size_t)tile * 16 * CS;
      #pragma unroll
      for (int r = 0; r < 8; ++r)
        s[ob + (size_t)(roff + r)*CS + col] = acc[j*4 + tile][r] + bias;
    }
  }
}

// ---------------------------------------------------------------------------
// Stage 3: gated TCN2 (16 -> 16ch) + per-node sum/sumsq for BatchNorm
// ---------------------------------------------------------------------------
__global__ __launch_bounds__(256) void k_tcn2(
    const float* __restrict__ s,                 // [32][22][1024][16]
    const float* __restrict__ w1, const float* __restrict__ b1,
    const float* __restrict__ w2, const float* __restrict__ b2,
    const float* __restrict__ w3, const float* __restrict__ b3,
    float* __restrict__ h2,                      // [32][20][1024][16]
    float* __restrict__ bnsum, float* __restrict__ bnsq)
{
  __shared__ float sw1[CS*CS*3], sw2[CS*CS*3], sw3[CS*CS*3], sb[3*CS];
  int tid = threadIdx.x;
  for (int i = tid; i < CS*CS*3; i += 256) { sw1[i]=w1[i]; sw2[i]=w2[i]; sw3[i]=w3[i]; }
  if (tid < CS) { sb[tid]=b1[tid]; sb[CS+tid]=b2[tid]; sb[2*CS+tid]=b3[tid]; }
  __syncthreads();

  int idx = blockIdx.x * 256 + tid;              // B*T2*N exact
  int n = idx & (NN - 1);
  int t = (idx >> 10) % T2;
  int b = idx / (NN * T2);

  const float* base = s + (((size_t)b*T1 + t)*NN + n)*CS;
  float v[3][CS];
  #pragma unroll
  for (int j = 0; j < 3; ++j) {
    const float4* p4 = reinterpret_cast<const float4*>(base + (size_t)j*NN*CS);
    #pragma unroll
    for (int q4 = 0; q4 < 4; ++q4) {
      float4 f = p4[q4];
      v[j][q4*4+0]=f.x; v[j][q4*4+1]=f.y; v[j][q4*4+2]=f.z; v[j][q4*4+3]=f.w;
    }
  }

  float lsum = 0.f, lsq = 0.f;
  float* outp = h2 + (((size_t)b*T2 + t)*NN + n)*CS;
  for (int o = 0; o < CS; ++o) {
    float p = sb[o], q = sb[CS+o], r = sb[2*CS+o];
    #pragma unroll
    for (int ci = 0; ci < CS; ++ci) {
      int wb = (o*CS + ci)*3;
      #pragma unroll
      for (int j = 0; j < 3; ++j) {
        float xv = v[j][ci];
        p += sw1[wb+j]*xv; q += sw2[wb+j]*xv; r += sw3[wb+j]*xv;
      }
    }
    float y = p * (1.f / (1.f + __expf(-q))) + tanhf(r);
    outp[o] = y;
    lsum += y; lsq += y*y;
  }
  atomicAdd(&bnsum[n], lsum);
  atomicAdd(&bnsq[n],  lsq);
}

// ---------------------------------------------------------------------------
// Stage 4: BatchNorm over (B,T',C) per node + ReLU
// ---------------------------------------------------------------------------
__global__ __launch_bounds__(256) void k_bn(
    const float* __restrict__ h2,
    const float* __restrict__ bnsum, const float* __restrict__ bnsq,
    const float* __restrict__ gamma, const float* __restrict__ beta,
    float* __restrict__ out)
{
  int idx = blockIdx.x * 256 + threadIdx.x;      // B*T2*N*CS exact
  int n = (idx >> 4) & (NN - 1);
  const float cnt = (float)(BB * T2 * CS);       // 10240
  float mean = bnsum[n] / cnt;
  float var  = bnsq[n] / cnt - mean*mean;
  float y = (h2[idx] - mean) * rsqrtf(var + 1e-5f) * gamma[n] + beta[n];
  out[idx] = fmaxf(y, 0.f);
}

__global__ __launch_bounds__(256) void k_zero(float* __restrict__ p, int nfloats) {
  int i = blockIdx.x * 256 + threadIdx.x;
  if (i < nfloats) p[i] = 0.f;
}

// ---------------------------------------------------------------------------
extern "C" void kernel_launch(void* const* d_in, const int* in_sizes, int n_in,
                              void* d_out, int out_size, void* d_ws, size_t ws_size,
                              hipStream_t stream) {
  const float* x    = (const float*)d_in[0];
  const float* cheb = (const float*)d_in[1];
  const float* t1w1 = (const float*)d_in[2];
  const float* t1b1 = (const float*)d_in[3];
  const float* t1w2 = (const float*)d_in[4];
  const float* t1b2 = (const float*)d_in[5];
  const float* t1w3 = (const float*)d_in[6];
  const float* t1b3 = (const float*)d_in[7];
  const float* wch  = (const float*)d_in[8];
  const float* bch  = (const float*)d_in[9];
  const float* t2w1 = (const float*)d_in[10];
  const float* t2b1 = (const float*)d_in[11];
  const float* t2w2 = (const float*)d_in[12];
  const float* t2b2 = (const float*)d_in[13];
  const float* t2w3 = (const float*)d_in[14];
  const float* t2b3 = (const float*)d_in[15];
  const float* gm   = (const float*)d_in[16];
  const float* bt   = (const float*)d_in[17];

  // workspace layout (bytes, 256-aligned offsets)
  char* ws = (char*)d_ws;
  _Float16* chebT = (_Float16*)(ws + 0);             //  6,291,456 B
  _Float16* hWt   = (_Float16*)(ws + 6291456);       // 69,206,016 B
  float*    sbuf  = (float*)   (ws + 75497472);      // 46,137,344 B
  float*    h2buf = (float*)   (ws + 121634816);     // 41,943,040 B
  float*    bnsum = (float*)   (ws + 163577856);     //      4,096 B
  float*    bnsq  = (float*)   (ws + 163577856 + 4096);

  k_chebT<<<12288, 256, 0, stream>>>(cheb, chebT);
  k_tcn1 <<<dim3(NN/256, T1, BB), 256, 0, stream>>>(
      x, t1w1, t1b1, t1w2, t1b2, t1w3, t1b3, wch, hWt);
  k_zero <<<8, 256, 0, stream>>>(bnsum, 2048);       // zeroes bnsum + bnsq
  k_cheb_wmma<<<dim3(2, (BB*T1)/2), 256, 0, stream>>>(chebT, hWt, bch, sbuf);
  k_tcn2 <<<(BB*T2*NN)/256, 256, 0, stream>>>(
      sbuf, t2w1, t2b1, t2w2, t2b2, t2w3, t2b3, h2buf, bnsum, bnsq);
  k_bn   <<<(BB*T2*NN*CS)/256, 256, 0, stream>>>(h2buf, bnsum, bnsq, gm, bt,
                                                 (float*)d_out);
}
